// MlpExpert_50036368998882
// MI455X (gfx1250) — matrix operand order repaired and checked by reading, not compiled, not run on hardware
//
#include <hip/hip_runtime.h>
#include <hip/hip_bf16.h>
#include <math.h>
#include <stdint.h>

// ---------------------------------------------------------------------------
// MoE MLP (grouped GEMM): fc1 + exact GELU -> H (bf16 in ws), then fc2 -> out.
// bf16 WMMA (v_wmma_f32_16x16x32_bf16), f32 accumulate.
// Block tile 256x128, k-step 32, 8 waves; each wave owns a 64x64 sub-tile
// -> 16 WMMAs per k-step from 8 fragment loads. Double-buffered LDS staging.
// Kernel-2 A staging uses GLOBAL_LOAD_ASYNC_TO_LDS_B128 (ASYNCcnt path).
// ---------------------------------------------------------------------------

typedef __bf16 bf16_t;
typedef __attribute__((ext_vector_type(16))) __bf16 v16bf;
typedef __attribute__((ext_vector_type(8)))  float  v8f;
typedef __attribute__((ext_vector_type(4)))  int    v4i;

typedef __attribute__((address_space(1))) v4i* as1_v4i_t;  // global int4*
typedef __attribute__((address_space(3))) v4i* as3_v4i_t;  // LDS int4*

#define D_IN   1024
#define D_HID  4096
#define D_OUT  1024
#define T_TOK  16384
#define CAP    2048   // tokens per expert

#define BM 256
#define BN 128
#define BK 32
#define LDT 40        // padded LDS row stride (elements): 80B -> conflict-free b128 reads

#if __has_builtin(__builtin_amdgcn_global_load_async_to_lds_b128)
#define HAVE_ASYNC_LDS 1
#else
#define HAVE_ASYNC_LDS 0
#endif

struct __align__(8) bf4 { bf16_t x, y, z, w; };

// ---- fragment builders (exact CDNA5 VGPR layouts, CDNA5 ISA) ----

// A-matrix 16x32 bf16: lane L holds row m = L%16.
// lanes 0-15:  VGPR0-3 = K 0..7,  VGPR4-7 = K 16..23
// lanes 16-31: VGPR0-3 = K 8..15, VGPR4-7 = K 24..31
__device__ inline v16bf frag_a(const bf16_t* tile, int mbase, int lane) {
  const int m  = mbase + (lane & 15);
  const int k0 = (lane & 16) ? 8 : 0;
  const bf16_t* p = tile + m * LDT + k0;
  union { v16bf v; uint4 q[2]; } u;
  u.q[0] = *(const uint4*)(p);        // K = k0 .. k0+7
  u.q[1] = *(const uint4*)(p + 16);   // K = k0+16 .. k0+23
  return u.v;
}

// B-matrix 32x16 bf16: lane L holds column n = L%16.
// lanes 0-15: K=0..7,16..23; lanes 16-31: K=8..15,24..31 (same as A).
// Source tile is stored N-major (row n = K-run), so this is two 16B runs.
__device__ inline v16bf frag_b(const bf16_t* tile, int nbase, int lane) {
  const int n  = nbase + (lane & 15);
  const int kb = (lane & 16) ? 8 : 0;
  const bf16_t* p = tile + n * LDT + kb;
  union { v16bf v; uint4 q[2]; } u;
  u.q[0] = *(const uint4*)(p);        // K = kb .. kb+7
  u.q[1] = *(const uint4*)(p + 16);   // K = kb+16 .. kb+23
  return u.v;
}

// ---- cooperative staging: ROWS x 32 cols into padded LDS tile ----

template <int ROWS>
__device__ inline void stage_f32(bf16_t* tile, const float* __restrict__ src,
                                 int srcStride, int tid) {
#pragma unroll
  for (int i = 0; i < (ROWS * 8) / 256; ++i) {
    const int idx = tid + i * 256;            // 8 float4 per row
    const int r = idx >> 3;
    const int c = (idx & 7) << 2;
    const float4 v = *(const float4*)(src + (size_t)r * srcStride + c);
    bf4 b;
    b.x = (bf16_t)v.x; b.y = (bf16_t)v.y; b.z = (bf16_t)v.z; b.w = (bf16_t)v.w;
    *(bf4*)(tile + r * LDT + c) = b;
  }
}

// bf16 -> bf16 pure copy: use CDNA5 async global->LDS DMA path when available.
template <int ROWS>
__device__ inline void stage_bf16(bf16_t* tile, const bf16_t* __restrict__ src,
                                  int srcStride, int tid) {
#pragma unroll
  for (int i = 0; i < (ROWS * 4) / 256; ++i) {
    const int idx = tid + i * 256;            // 4 x uint4 per row
    const int r = idx >> 2;
    const int c = (idx & 3) << 3;
    bf16_t* dst = tile + r * LDT + c;
    const bf16_t* s = src + (size_t)r * srcStride + c;
#if HAVE_ASYNC_LDS
    __builtin_amdgcn_global_load_async_to_lds_b128(
        (as1_v4i_t)(uintptr_t)s,
        (as3_v4i_t)(unsigned)(uintptr_t)dst,
        0, 0);
#else
    *(uint4*)dst = *(const uint4*)s;
#endif
  }
}

__device__ inline void async_fence() {
#if HAVE_ASYNC_LDS
#if __has_builtin(__builtin_amdgcn_s_wait_asynccnt)
  __builtin_amdgcn_s_wait_asynccnt(0);
#else
  asm volatile("s_wait_asynccnt 0x0" ::: "memory");
#endif
#endif
}

__device__ inline v8f wmma_bf16(v16bf a, v16bf b, v8f c) {
  return __builtin_amdgcn_wmma_f32_16x16x32_bf16(false, a, false, b,
                                                 (short)0, c, false, false);
}

// ---------------------------------------------------------------------------
// Kernel 1: H = gelu(X @ W1^T + b1), per expert.  X f32 [T,1024],
// W1 f32 [E,4096,1024] (N-major, K contiguous) -> H bf16 [T,4096].
// ---------------------------------------------------------------------------
__global__ void __launch_bounds__(256)
moe_fc1_gelu(const float* __restrict__ x, const float* __restrict__ w1,
             const float* __restrict__ b1, bf16_t* __restrict__ h) {
  __shared__ __align__(16) bf16_t Abuf[2][BM * LDT];
  __shared__ __align__(16) bf16_t Bbuf[2][BN * LDT];

  const int tid  = threadIdx.x;
  const int lane = tid & 31;
  const int wave = tid >> 5;
  const int wm   = wave >> 1;   // 0..3 -> 64-row strip
  const int wn   = wave & 1;    // 0..1 -> 64-col strip

  const int rowbase = blockIdx.y * BM;
  const int colbase = blockIdx.x * BN;
  const int e = rowbase / CAP;  // BM divides CAP: tiles never cross experts

  const float* Asrc = x  + (size_t)rowbase * D_IN;
  const float* Bsrc = w1 + (size_t)e * D_HID * D_IN + (size_t)colbase * D_IN;

  v8f acc[4][4];
#pragma unroll
  for (int mt = 0; mt < 4; ++mt)
#pragma unroll
    for (int nt = 0; nt < 4; ++nt) acc[mt][nt] = (v8f){0,0,0,0,0,0,0,0};

  stage_f32<BM>(Abuf[0], Asrc, D_IN, tid);
  stage_f32<BN>(Bbuf[0], Bsrc, D_IN, tid);
  __syncthreads();

  // per-thread prefetch targets (rows of next-next tiles)
  const char* pfA = (const char*)(Asrc + (size_t)tid * D_IN);
  const char* pfB = (const char*)(Bsrc + (size_t)(tid & 127) * D_IN);

  const int KT = D_IN / BK;  // 32
  for (int kt = 0; kt < KT; ++kt) {
    const int cur = kt & 1;
    if (kt + 2 < KT) {
      __builtin_prefetch(pfA + (size_t)(kt + 2) * BK * 4, 0, 0);
      __builtin_prefetch(pfB + (size_t)(kt + 2) * BK * 4, 0, 0);
    }
    if (kt + 1 < KT) {
      stage_f32<BM>(Abuf[cur ^ 1], Asrc + (size_t)(kt + 1) * BK, D_IN, tid);
      stage_f32<BN>(Bbuf[cur ^ 1], Bsrc + (size_t)(kt + 1) * BK, D_IN, tid);
    }
    v16bf af[4], bf[4];
#pragma unroll
    for (int mt = 0; mt < 4; ++mt) af[mt] = frag_a(Abuf[cur], wm * 64 + mt * 16, lane);
#pragma unroll
    for (int nt = 0; nt < 4; ++nt) bf[nt] = frag_b(Bbuf[cur], wn * 64 + nt * 16, lane);
#pragma unroll
    for (int mt = 0; mt < 4; ++mt)
#pragma unroll
      for (int nt = 0; nt < 4; ++nt)
        acc[mt][nt] = wmma_bf16(af[mt], bf[nt], acc[mt][nt]);
    __syncthreads();
  }

  // epilogue: bias + exact-erf GELU, store bf16.  C/D layout: lane=N, VGPR=M.
  const int nlo = lane & 15;
  const int mhi = (lane >> 4) * 8;
#pragma unroll
  for (int mt = 0; mt < 4; ++mt) {
#pragma unroll
    for (int nt = 0; nt < 4; ++nt) {
      const int col = colbase + wn * 64 + nt * 16 + nlo;
      const int row = rowbase + wm * 64 + mt * 16 + mhi;
      const float bias = b1[(size_t)e * D_HID + col];
      bf16_t* dst = h + (size_t)row * D_HID + col;
#pragma unroll
      for (int v = 0; v < 8; ++v) {
        const float t = acc[mt][nt][v] + bias;
        const float g = 0.5f * t * (1.0f + erff(t * 0.70710678118654752f));
        dst[(size_t)v * D_HID] = (bf16_t)g;
      }
    }
  }
}

// ---------------------------------------------------------------------------
// Kernel 2: out = H @ W2^T + b2, per expert.  H bf16 [T,4096],
// W2 f32 [E,1024,4096] (N-major, K contiguous) -> out f32 [T,1024].
// ---------------------------------------------------------------------------
__global__ void __launch_bounds__(256)
moe_fc2(const bf16_t* __restrict__ h, const float* __restrict__ w2,
        const float* __restrict__ b2, float* __restrict__ out) {
  __shared__ __align__(16) bf16_t Abuf[2][BM * LDT];
  __shared__ __align__(16) bf16_t Bbuf[2][BN * LDT];

  const int tid  = threadIdx.x;
  const int lane = tid & 31;
  const int wave = tid >> 5;
  const int wm   = wave >> 1;
  const int wn   = wave & 1;

  const int rowbase = blockIdx.y * BM;
  const int colbase = blockIdx.x * BN;
  const int e = rowbase / CAP;

  const bf16_t* Asrc = h  + (size_t)rowbase * D_HID;
  const float*  Bsrc = w2 + (size_t)e * D_OUT * D_HID + (size_t)colbase * D_HID;

  v8f acc[4][4];
#pragma unroll
  for (int mt = 0; mt < 4; ++mt)
#pragma unroll
    for (int nt = 0; nt < 4; ++nt) acc[mt][nt] = (v8f){0,0,0,0,0,0,0,0};

  stage_bf16<BM>(Abuf[0], Asrc, D_HID, tid);
  stage_f32 <BN>(Bbuf[0], Bsrc, D_HID, tid);
  async_fence();
  __syncthreads();

  const char* pfA = (const char*)(Asrc + (size_t)tid * D_HID);
  const char* pfB = (const char*)(Bsrc + (size_t)(tid & 127) * D_HID);

  const int KT = D_HID / BK;  // 128
  for (int kt = 0; kt < KT; ++kt) {
    const int cur = kt & 1;
    if (kt + 2 < KT) {
      __builtin_prefetch(pfA + (size_t)(kt + 2) * BK * 2, 0, 0);
      __builtin_prefetch(pfB + (size_t)(kt + 2) * BK * 4, 0, 0);
    }
    if (kt + 1 < KT) {
      stage_bf16<BM>(Abuf[cur ^ 1], Asrc + (size_t)(kt + 1) * BK, D_HID, tid);
      stage_f32 <BN>(Bbuf[cur ^ 1], Bsrc + (size_t)(kt + 1) * BK, D_HID, tid);
    }
    v16bf af[4], bf[4];
#pragma unroll
    for (int mt = 0; mt < 4; ++mt) af[mt] = frag_a(Abuf[cur], wm * 64 + mt * 16, lane);
#pragma unroll
    for (int nt = 0; nt < 4; ++nt) bf[nt] = frag_b(Bbuf[cur], wn * 64 + nt * 16, lane);
#pragma unroll
    for (int mt = 0; mt < 4; ++mt)
#pragma unroll
      for (int nt = 0; nt < 4; ++nt)
        acc[mt][nt] = wmma_bf16(af[mt], bf[nt], acc[mt][nt]);
    async_fence();
    __syncthreads();
  }

  const int nlo = lane & 15;
  const int mhi = (lane >> 4) * 8;
#pragma unroll
  for (int mt = 0; mt < 4; ++mt) {
#pragma unroll
    for (int nt = 0; nt < 4; ++nt) {
      const int col = colbase + wn * 64 + nt * 16 + nlo;
      const int row = rowbase + wm * 64 + mt * 16 + mhi;
      const float bias = b2[(size_t)e * D_OUT + col];
      float* dst = out + (size_t)row * D_OUT + col;
#pragma unroll
      for (int v = 0; v < 8; ++v) {
        dst[(size_t)v * D_OUT] = acc[mt][nt][v] + bias;
      }
    }
  }
}

// ---------------------------------------------------------------------------
extern "C" void kernel_launch(void* const* d_in, const int* in_sizes, int n_in,
                              void* d_out, int out_size, void* d_ws, size_t ws_size,
                              hipStream_t stream) {
  const float* x  = (const float*)d_in[0];
  // d_in[1] = cnt (int32[E]) — tokens are an equal contiguous split, unused.
  const float* w1 = (const float*)d_in[2];
  const float* b1 = (const float*)d_in[3];
  const float* w2 = (const float*)d_in[4];
  const float* b2 = (const float*)d_in[5];
  float*  out = (float*)d_out;
  bf16_t* H   = (bf16_t*)d_ws;   // 16384 x 4096 bf16 = 128 MB scratch

  dim3 blk(256);
  dim3 g1(D_HID / BN, T_TOK / BM);  // 32 x 64
  dim3 g2(D_OUT / BN, T_TOK / BM);  //  8 x 64

  moe_fc1_gelu<<<g1, blk, 0, stream>>>(x, w1, b1, H);
  moe_fc2    <<<g2, blk, 0, stream>>>(H, w2, b2, out);
}
